// NormalVitBlock_24644522344902
// MI455X (gfx1250) — compile-verified, matrix operand order checked by reading, not run on hardware
//
#include <hip/hip_runtime.h>

typedef __attribute__((ext_vector_type(16))) __bf16 v16bf;
typedef __attribute__((ext_vector_type(8)))  float  v8f;

#define HDIM 128
#define WDIM 128
#define CD   256
#define HWT  16384

static __device__ __forceinline__ v8f bwmma(v16bf a, v16bf b, v8f c) {
  // D = A(16x32 bf16) * B(32x16 bf16) + C(16x16 f32)
  return __builtin_amdgcn_wmma_f32_16x16x32_bf16(false, a, false, b, (short)0, c, false, false);
}

// A operand: row-major source [M][K], tile 16x32. Lane<16 holds row m=lane,
// K 0..7 then 16..23; lane>=16 holds row m=lane-16, K 8..15 then 24..31.
// Also used for B operands whose source is stored K-contiguous (W^T, K^T, V^T):
// lane then holds column n of B with the same K split.
static __device__ __forceinline__ v16bf lda_bf(const __bf16* s, int ld, int lane) {
  int m = lane & 15, kb = (lane >> 4) << 3;
  v16bf a;
#pragma unroll
  for (int i = 0; i < 8; ++i) {
    a[i]     = s[m * ld + kb + i];
    a[i + 8] = s[m * ld + kb + 16 + i];
  }
  return a;
}

static __device__ __forceinline__ v16bf lda_f32(const float* s, int ld, int lane) {
  int m = lane & 15, kb = (lane >> 4) << 3;
  v16bf a;
#pragma unroll
  for (int i = 0; i < 8; ++i) {
    a[i]     = (__bf16)s[m * ld + kb + i];
    a[i + 8] = (__bf16)s[m * ld + kb + 16 + i];
  }
  return a;
}

// C/D layout: VGPR r -> row (lane<16 ? r : 8+r), col = lane&15
static __device__ __forceinline__ void st_c_f32(float* d, int ld, int lane, v8f c) {
  int n = lane & 15, mo = (lane >> 4) << 3;
#pragma unroll
  for (int r = 0; r < 8; ++r) d[(mo + r) * ld + n] = c[r];
}

static __device__ __forceinline__ void st_c_bf(__bf16* d, int ld, int lane, v8f c) {
  int n = lane & 15, mo = (lane >> 4) << 3;
#pragma unroll
  for (int r = 0; r < 8; ++r) d[(mo + r) * ld + n] = (__bf16)c[r];
}

// transposed store: d is [col][row] with leading dim ld over rows
static __device__ __forceinline__ void st_c_bf_T(__bf16* d, int ld, int lane, v8f c) {
  int n = lane & 15, mo = (lane >> 4) << 3;
#pragma unroll
  for (int r = 0; r < 8; ++r) d[n * ld + mo + r] = (__bf16)c[r];
}

// ---------------- kernel 0: weights fp32 -> bf16, stored TRANSPOSED [N][K] ----------------
__global__ __launch_bounds__(256) void k_cvt_weights(const float* __restrict__ ql,
                                                     const float* __restrict__ kvl,
                                                     const float* __restrict__ qg,
                                                     const float* __restrict__ kvg,
                                                     const float* __restrict__ pj,
                                                     __bf16* __restrict__ wbf) {
  __bf16* dql  = wbf;            // [256][256]
  __bf16* dkvl = wbf + 65536;    // [512][256]
  __bf16* dqg  = wbf + 196608;   // [256][256]
  __bf16* dkvg = wbf + 262144;   // [512][256]
  __bf16* dpj  = wbf + 393216;   // [256][256]
  int i = blockIdx.x * 256 + threadIdx.x;
  if (i < 65536)        { int k = i >> 8, n = i & 255;               dql[n * 256 + k]  = (__bf16)ql[i]; }
  else if (i < 196608)  { int j = i - 65536;  int k = j >> 9, n = j & 511; dkvl[n * 256 + k] = (__bf16)kvl[j]; }
  else if (i < 262144)  { int j = i - 196608; int k = j >> 8, n = j & 255; dqg[n * 256 + k]  = (__bf16)qg[j]; }
  else if (i < 393216)  { int j = i - 262144; int k = j >> 9, n = j & 511; dkvg[n * 256 + k] = (__bf16)kvg[j]; }
  else if (i < 458752)  { int j = i - 393216; int k = j >> 8, n = j & 255; dpj[n * 256 + k]  = (__bf16)pj[j]; }
}

// ---------------- kernel 1: local windowed attention ----------------
__global__ __launch_bounds__(256) void k_local_attn(const float* __restrict__ fm,
                                                    const __bf16* __restrict__ wqlT,
                                                    const __bf16* __restrict__ wkvlT,
                                                    __bf16* __restrict__ winx) {
  __shared__ __bf16 Xs[64 * CD];   // 32 KB  (token-major)
  __shared__ __bf16 Qs[64 * CD];   // token-major
  __shared__ __bf16 Ks[64 * CD];   // token-major (used as K^T B-operand)
  __shared__ __bf16 Vt[CD * 64];   // 32 KB  V TRANSPOSED: [ch][token]
  __shared__ float  Ss[64 * 64];   // 16 KB   (total 144 KB < 320 KB/WGP)
  const int tid = threadIdx.x, wave = tid >> 5, lane = tid & 31;
  const int wb = blockIdx.x, b = wb >> 8, wi = wb & 255;
  const int bh = wi >> 4, bw = wi & 15;

  __builtin_prefetch(wqlT, 0, 0);
  __builtin_prefetch(wkvlT, 0, 0);

  // gather window tile (64 tokens x 256 ch) from NCHW feature map
  for (int idx = tid; idx < 64 * CD; idx += 256) {
    int j = idx >> 8, c = idx & 255;
    int h = (bh << 3) + (j >> 3), w = (bw << 3) + (j & 7);
    Xs[idx] = (__bf16)fm[(((size_t)b * CD + c) * HDIM + h) * WDIM + w];
  }
  __syncthreads();

  // Q = X @ ql (64x256) and KV = X @ kvl (64x512), 192 output tiles / 8 waves
  for (int t = wave; t < 192; t += 8) {
    v8f acc = {};
    if (t < 64) {
      int mt = t >> 4, nt = t & 15;
      for (int kk = 0; kk < CD; kk += 32) {
        v16bf a  = lda_bf(Xs + mt * 16 * CD + kk, CD, lane);
        v16bf bb = lda_bf(wqlT + (size_t)(nt * 16) * CD + kk, CD, lane);  // W^T, K-contig
        acc = bwmma(a, bb, acc);
      }
      st_c_bf(Qs + mt * 16 * CD + nt * 16, CD, lane, acc);
    } else {
      int t2 = t - 64, mt = t2 >> 5, nt = t2 & 31, nc = nt * 16;
      for (int kk = 0; kk < CD; kk += 32) {
        v16bf a  = lda_bf(Xs + mt * 16 * CD + kk, CD, lane);
        v16bf bb = lda_bf(wkvlT + (size_t)nc * CD + kk, CD, lane);        // W^T, K-contig
        acc = bwmma(a, bb, acc);
      }
      if (nc < 256) st_c_bf(Ks + mt * 16 * CD + nc, CD, lane, acc);
      else          st_c_bf_T(Vt + (size_t)(nc - 256) * 64 + mt * 16, 64, lane, acc);
    }
  }
  __syncthreads();

  const float scale = 0.17677669529663687f;  // 32^-0.5
  for (int hd = 0; hd < 8; ++hd) {
    // S = Q_h (64x32) @ K_h^T (32x64): 16 tiles, single K-step
    for (int t = wave; t < 16; t += 8) {
      int mt = t >> 2, nt = t & 3;
      v16bf a  = lda_bf(Qs + mt * 16 * CD + hd * 32, CD, lane);
      v16bf bk = lda_bf(Ks + nt * 16 * CD + hd * 32, CD, lane);  // K^T via A-style load
      v8f acc = {};
      acc = bwmma(a, bk, acc);
      st_c_f32(Ss + mt * 16 * 64 + nt * 16, 64, lane, acc);
    }
    __syncthreads();
    if (tid < 64) {  // softmax rows
      float* row = Ss + tid * 64;
      float mx = -1e30f;
      for (int i = 0; i < 64; ++i) mx = fmaxf(mx, row[i] * scale);
      float sm = 0.f;
      for (int i = 0; i < 64; ++i) { float e = __expf(row[i] * scale - mx); row[i] = e; sm += e; }
      float inv = 1.f / sm;
      for (int i = 0; i < 64; ++i) row[i] *= inv;
    }
    __syncthreads();
    {  // out_h = P (64x64) @ V_h (64x32): 8 tiles, one per wave; fuse residual
      int mt = wave >> 1, nt = wave & 1;
      v8f acc = {};
#pragma unroll
      for (int kk = 0; kk < 64; kk += 32) {
        v16bf a  = lda_f32(Ss + mt * 16 * 64 + kk, 64, lane);
        v16bf bv = lda_bf(Vt + (size_t)(hd * 32 + nt * 16) * 64 + kk, 64, lane);  // V^T, K-contig
        acc = bwmma(a, bv, acc);
      }
      int n = lane & 15, mo = (lane >> 4) << 3;
#pragma unroll
      for (int r = 0; r < 8; ++r) {
        int m = mt * 16 + mo + r, cc = hd * 32 + nt * 16 + n;
        float v = acc[r] + (float)Xs[m * CD + cc];
        winx[((size_t)wb * 64 + m) * CD + cc] = (__bf16)v;
      }
    }
    __syncthreads();
  }
}

// ---------------- kernel 2a: depthwise SR conv + LN + GELU ----------------
__global__ __launch_bounds__(256) void k_sr_gx(const float* __restrict__ fm,
                                               const float* __restrict__ srw,
                                               const float* __restrict__ srb,
                                               const float* __restrict__ nw,
                                               const float* __restrict__ nb,
                                               __bf16* __restrict__ gx) {
  __shared__ float red[256];
  __shared__ float red2[256];
  const int tid = threadIdx.x;
  const int g = blockIdx.x, b = g >> 8, pt = g & 255;
  const int py = pt >> 4, px = pt & 15;
  const int c = tid;
  const float* base = fm + (((size_t)b * CD + c) * HDIM + py * 8) * WDIM + px * 8;
  float s = 0.f;
#pragma unroll
  for (int i = 0; i < 8; ++i)
#pragma unroll
    for (int j = 0; j < 8; ++j)
      s += base[i * WDIM + j] * srw[c * 64 + i * 8 + j];
  s += srb[c];
  red[tid] = s; red2[tid] = s * s;
  __syncthreads();
  for (int off = 128; off > 0; off >>= 1) {
    if (tid < off) { red[tid] += red[tid + off]; red2[tid] += red2[tid + off]; }
    __syncthreads();
  }
  float m = red[0] * (1.f / 256.f);
  float var = red2[0] * (1.f / 256.f) - m * m;
  float xn = (s - m) * rsqrtf(var + 1e-5f) * nw[c] + nb[c];
  float ge = 0.5f * xn * (1.f + erff(xn * 0.70710678118654752f));  // exact GELU
  gx[(size_t)g * CD + c] = (__bf16)ge;
}

// ---------------- kernel 2b: kv_g = gx @ kvg_w (K row-major, V transposed) ----------------
__global__ __launch_bounds__(256) void k_kvg_gemm(const __bf16* __restrict__ gx,
                                                  const __bf16* __restrict__ wkvgT,
                                                  __bf16* __restrict__ Kg,    // [B*256 tok][256 ch]
                                                  __bf16* __restrict__ VgT) { // [B][256 ch][256 tok]
  __shared__ __bf16 As[64 * CD];
  const int tid = threadIdx.x, wave = tid >> 5, lane = tid & 31;
  const int mb = blockIdx.x, nb = blockIdx.y;  // 16 x 2 blocks
  // vectorized 16B copy of the A tile
  {
    const uint4* src = (const uint4*)(gx + (size_t)mb * 64 * CD);
    uint4* dst = (uint4*)As;
    for (int i = tid; i < (64 * CD) / 8; i += 256) dst[i] = src[i];
  }
  __syncthreads();
  for (int t = wave; t < 64; t += 8) {
    int mt = t >> 4, nt = t & 15;
    int ng = nb * 256 + nt * 16;
    v8f acc = {};
    for (int kk = 0; kk < CD; kk += 32) {
      v16bf a  = lda_bf(As + mt * 16 * CD + kk, CD, lane);
      v16bf bb = lda_bf(wkvgT + (size_t)ng * CD + kk, CD, lane);  // W^T, K-contig
      acc = bwmma(a, bb, acc);
    }
    int row0 = mb * 64 + mt * 16;
    if (ng < 256) {
      st_c_bf(Kg + (size_t)row0 * CD + ng, CD, lane, acc);
    } else {
      int bb2 = row0 >> 8, tok0 = row0 & 255;
      st_c_bf_T(VgT + (size_t)bb2 * 65536 + (size_t)(ng - 256) * 256 + tok0, 256, lane, acc);
    }
  }
}

// ---------------- kernel 3: global attention + residual quirk + proj ----------------
__global__ __launch_bounds__(256) void k_global_attn(const __bf16* __restrict__ winx,
                                                     const __bf16* __restrict__ Kg,
                                                     const __bf16* __restrict__ VgT,
                                                     const __bf16* __restrict__ wqgT,
                                                     const __bf16* __restrict__ wprojT,
                                                     const float* __restrict__ pb,
                                                     float* __restrict__ out) {
  __shared__ __bf16 Xs[64 * CD];   // q source; reused as pre-proj activations
  __shared__ __bf16 Qs[64 * CD];
  __shared__ float  Ss[64 * 256];  // scores per head (128 KB total LDS)
  const int tid = threadIdx.x, wave = tid >> 5, lane = tid & 31;
  const int g = blockIdx.x, b = g >> 8, tg = g & 255;

  __builtin_prefetch(wqgT, 0, 0);
  __builtin_prefetch(wprojT, 0, 0);

  {  // vectorized 16B copy; window-order rows of winx are contiguous
    const uint4* src = (const uint4*)(winx + (size_t)g * 64 * CD);
    uint4* dst = (uint4*)Xs;
    for (int i = tid; i < (64 * CD) / 8; i += 256) dst[i] = src[i];
  }
  __syncthreads();

  // Q = X @ qg
  for (int t = wave; t < 64; t += 8) {
    int mt = t >> 4, nt = t & 15;
    v8f acc = {};
    for (int kk = 0; kk < CD; kk += 32) {
      v16bf a  = lda_bf(Xs + mt * 16 * CD + kk, CD, lane);
      v16bf bb = lda_bf(wqgT + (size_t)(nt * 16) * CD + kk, CD, lane);  // W^T, K-contig
      acc = bwmma(a, bb, acc);
    }
    st_c_bf(Qs + mt * 16 * CD + nt * 16, CD, lane, acc);
  }
  __syncthreads();

  __bf16* X2 = Xs;  // Xs free after Q GEMM
  const float scale = 0.17677669529663687f;

  for (int hd = 0; hd < 8; ++hd) {
    // S = Q_h (64x32) @ Kg^T (32x256): 64 tiles, 1 K-step
    for (int t = wave; t < 64; t += 8) {
      int mt = t >> 4, nt = t & 15;
      v16bf a  = lda_bf(Qs + mt * 16 * CD + hd * 32, CD, lane);
      v16bf bk = lda_bf(Kg + ((size_t)b * 256 + nt * 16) * CD + hd * 32, CD, lane);  // K^T, ch-contig
      v8f acc = {};
      acc = bwmma(a, bk, acc);
      st_c_f32(Ss + mt * 16 * 256 + nt * 16, 256, lane, acc);
    }
    __syncthreads();
    if (tid < 64) {  // softmax over 256 keys
      float* row = Ss + tid * 256;
      float mx = -1e30f;
      for (int i = 0; i < 256; ++i) mx = fmaxf(mx, row[i] * scale);
      float sm = 0.f;
      for (int i = 0; i < 256; ++i) { float e = __expf(row[i] * scale - mx); row[i] = e; sm += e; }
      float inv = 1.f / sm;
      for (int i = 0; i < 256; ++i) row[i] *= inv;
    }
    __syncthreads();
    {  // out_h = P (64x256) @ V_h (256x32): one tile/wave, fuse spatial-reorder residual
      int mt = wave >> 1, nt = wave & 1;
      v8f acc = {};
      for (int kk = 0; kk < 256; kk += 32) {
        v16bf a  = lda_f32(Ss + mt * 16 * 256 + kk, 256, lane);
        v16bf bv = lda_bf(VgT + (size_t)b * 65536 + (size_t)(hd * 32 + nt * 16) * 256 + kk,
                          256, lane);  // V^T, K(token)-contig
        acc = bwmma(a, bv, acc);
      }
      int n = lane & 15, mo = (lane >> 4) << 3;
#pragma unroll
      for (int r = 0; r < 8; ++r) {
        int m = mt * 16 + mo + r, cc = hd * 32 + nt * 16 + n;
        int tl = tg * 64 + m;                 // spatial flat index within batch
        int hs = tl >> 7, wsp = tl & 127;
        int u = (((hs >> 3) << 4) + (wsp >> 3)) * 64 + ((hs & 7) << 3) + (wsp & 7);
        float resid = (float)winx[((size_t)b * HWT + u) * CD + cc];
        X2[m * CD + cc] = (__bf16)(acc[r] + resid);
      }
    }
    __syncthreads();
  }

  // final proj: out = X2 @ proj_w + proj_b, scattered to NCHW output
  for (int t = wave; t < 64; t += 8) {
    int mt = t >> 4, nt = t & 15;
    v8f acc = {};
    for (int kk = 0; kk < CD; kk += 32) {
      v16bf a  = lda_bf(X2 + mt * 16 * CD + kk, CD, lane);
      v16bf bb = lda_bf(wprojT + (size_t)(nt * 16) * CD + kk, CD, lane);  // W^T, K-contig
      acc = bwmma(a, bb, acc);
    }
    int n = lane & 15, mo = (lane >> 4) << 3;
#pragma unroll
    for (int r = 0; r < 8; ++r) {
      int m = mt * 16 + mo + r, cc = nt * 16 + n;
      int tl = tg * 64 + m;
      out[((size_t)b * CD + cc) * HWT + tl] = acc[r] + pb[cc];
    }
  }
}

// ---------------- host launcher ----------------
extern "C" void kernel_launch(void* const* d_in, const int* in_sizes, int n_in,
                              void* d_out, int out_size, void* d_ws, size_t ws_size,
                              hipStream_t stream) {
  (void)in_sizes; (void)n_in; (void)out_size; (void)ws_size;
  const float* fm   = (const float*)d_in[0];
  // d_in[1] (uncertain_map) is unused by the reference
  const float* sr_w = (const float*)d_in[2];
  const float* sr_b = (const float*)d_in[3];
  const float* nw   = (const float*)d_in[4];
  const float* nb   = (const float*)d_in[5];
  const float* ql   = (const float*)d_in[6];
  const float* kvl  = (const float*)d_in[7];
  const float* qg   = (const float*)d_in[8];
  const float* kvg  = (const float*)d_in[9];
  const float* pw   = (const float*)d_in[10];
  const float* pb   = (const float*)d_in[11];
  float* out = (float*)d_out;

  char* ws = (char*)d_ws;
  __bf16* wbf    = (__bf16*)ws;                    // 458752 bf16 transposed weights
  __bf16* wqlT   = wbf;
  __bf16* wkvlT  = wbf + 65536;
  __bf16* wqgT   = wbf + 196608;
  __bf16* wkvgT  = wbf + 262144;
  __bf16* wprojT = wbf + 393216;
  __bf16* winx   = (__bf16*)(ws + ((size_t)1  << 20));           // 32 MB: B*HWT*C bf16
  __bf16* gx     = (__bf16*)(ws + ((size_t)33 << 20));           // 0.5 MB
  __bf16* Kg     = (__bf16*)(ws + ((size_t)34 << 20));           // 0.5 MB [B*256][256]
  __bf16* VgT    = (__bf16*)(ws + ((size_t)34 << 20) + (1 << 19)); // 0.5 MB [B][256][256]

  k_cvt_weights<<<1792, 256, 0, stream>>>(ql, kvl, qg, kvg, pw, wbf);
  k_sr_gx      <<<1024, 256, 0, stream>>>(fm, sr_w, sr_b, nw, nb, gx);
  k_local_attn <<<1024, 256, 0, stream>>>(fm, wqlT, wkvlT, winx);
  k_kvg_gemm   <<<dim3(16, 2), 256, 0, stream>>>(gx, wkvgT, Kg, VgT);
  k_global_attn<<<1024, 256, 0, stream>>>(winx, Kg, VgT, wqgT, wprojT, pb, out);
}
